// ConditionedS5Transition_30305289240741
// MI455X (gfx1250) — matrix operand order repaired and checked by reading, not compiled, not run on hardware
//
#include <hip/hip_runtime.h>
#include <hip/hip_bf16.h>
#include <math.h>

typedef __bf16 bf16;
typedef __attribute__((ext_vector_type(16))) __bf16 v16bf;
typedef __attribute__((ext_vector_type(8)))  __bf16 v8bf;
typedef __attribute__((ext_vector_type(8)))  float  v8f;

#define B_   1024
#define T_   256
#define D_   128
#define HZ_  384
#define NOBS_ 25
#define UD_  16

// ---------------------------------------------------------------------------
// WMMA helpers (CDNA5 gfx1250, wave32, 16x16x32 bf16 -> f32)
// ---------------------------------------------------------------------------
__device__ __forceinline__ v8f vzero8() {
  v8f z;
#pragma unroll
  for (int i = 0; i < 8; ++i) z[i] = 0.0f;
  return z;
}

__device__ __forceinline__ v8f wmma_bf16(v16bf a, v16bf b, v8f c) {
  // (neg_a, A, neg_b, B, c_mod, C, reuse_a, reuse_b)
  return __builtin_amdgcn_wmma_f32_16x16x32_bf16(false, a, false, b, (short)0, c,
                                                 false, false);
}

// A-fragment (16xK row-major bf16 tile in LDS). Per ISA 7.12.2:
// lane l (0-15): row m=l, k = k0 + {0..7, 16..23}; lane l+16: k = k0 + {8..15, 24..31}
__device__ __forceinline__ v16bf a_frag(const bf16* base, int stride, int k0, int lane) {
  const int m  = lane & 15;
  const int kb = (lane & 16) >> 1;  // 0 or 8
  const bf16* p = base + m * stride + k0 + kb;
  v8bf lo = *(const v8bf*)(p);
  v8bf hi = *(const v8bf*)(p + 16);
  return __builtin_shufflevector(lo, hi, 0, 1, 2, 3, 4, 5, 6, 7,
                                 8, 9, 10, 11, 12, 13, 14, 15);
}

// B-fragment: pre-swizzled in global ws so each lane does ONE contiguous 32B load.
__device__ __forceinline__ v16bf b_frag(const bf16* packed, int kt, int nt, int NT,
                                        int lane) {
  return *(const v16bf*)(packed + (((size_t)kt * NT + nt) * 32 + lane) * 16);
}

// ---------------------------------------------------------------------------
// Prep kernel 1: fold BatchNorm scale/bias into linear layers
// ---------------------------------------------------------------------------
__global__ void fold_bn_k(const float* __restrict__ b, const float* __restrict__ g,
                          const float* __restrict__ bb, const float* __restrict__ m,
                          const float* __restrict__ v, float* __restrict__ s_out,
                          float* __restrict__ bias_out) {
  int i = threadIdx.x;
  if (i < 128) {
    float sc = g[i] * rsqrtf(v[i] + 1e-5f);
    s_out[i] = sc;
    bias_out[i] = (b[i] - m[i]) * sc + bb[i];
  }
}

// ---------------------------------------------------------------------------
// Prep kernel 2: pack a row-major [N][K] f32 weight matrix (optionally scaled
// per output row) into the WMMA B-operand swizzle:  dst[((kt*NT+nt)*32+l)*16+j]
//   = W[n][k],  n = nt*16 + (l&15),  k = kt*32 + ((l&16)?8:0) + (j&7) + ((j&8)?16:0)
// ---------------------------------------------------------------------------
__global__ void pack_b_k(const float* __restrict__ w, const float* __restrict__ scale,
                         bf16* __restrict__ dst, int N, int K, int KT, int NT) {
  int total = KT * NT * 512;
  for (int idx = blockIdx.x * blockDim.x + threadIdx.x; idx < total;
       idx += gridDim.x * blockDim.x) {
    int j = idx & 15;
    int l = (idx >> 4) & 31;
    int tile = idx >> 9;
    int nt = tile % NT;
    int kt = tile / NT;
    int n = nt * 16 + (l & 15);
    int k = kt * 32 + ((l & 16) ? 8 : 0) + (j & 7) + ((j & 8) ? 16 : 0);
    float val = 0.0f;
    if (k < K && n < N) {
      val = w[(size_t)n * K + k];
      if (scale) val *= scale[n];
    }
    dst[idx] = (bf16)val;
  }
}

// ---------------------------------------------------------------------------
// Prep kernel 3: 128x128 complex Gauss-Jordan inversion of V (single block).
// ---------------------------------------------------------------------------
__global__ void invert_k(const float* __restrict__ Vr, const float* __restrict__ Vi,
                         float* __restrict__ Ar, float* __restrict__ Ai,
                         float* __restrict__ Mr, float* __restrict__ Mi) {
  const int n = 128;
  const int tid = threadIdx.x;
  __shared__ float Fr[128], Fi[128];
  for (int i = tid; i < n * n; i += 256) {
    Ar[i] = Vr[i]; Ai[i] = Vi[i];
    Mr[i] = ((i / n) == (i % n)) ? 1.0f : 0.0f;
    Mi[i] = 0.0f;
  }
  __syncthreads();
#pragma unroll 1
  for (int p = 0; p < n; ++p) {
    float pr = Ar[p * n + p], pi = Ai[p * n + p];
    float den = pr * pr + pi * pi;
    den = den > 1e-30f ? den : 1e-30f;
    float ir = pr / den, ii = -pi / den;
    __syncthreads();
    for (int c = tid; c < n; c += 256) {   // scale pivot row
      float ar = Ar[p * n + c], ai = Ai[p * n + c];
      Ar[p * n + c] = ar * ir - ai * ii;
      Ai[p * n + c] = ar * ii + ai * ir;
      float mr = Mr[p * n + c], mi = Mi[p * n + c];
      Mr[p * n + c] = mr * ir - mi * ii;
      Mi[p * n + c] = mr * ii + mi * ir;
    }
    __syncthreads();
    for (int r = tid; r < n; r += 256) {   // capture factors before elimination
      if (r == p) { Fr[r] = 0.0f; Fi[r] = 0.0f; }
      else        { Fr[r] = Ar[r * n + p]; Fi[r] = Ai[r * n + p]; }
    }
    __syncthreads();
    for (int idx = tid; idx < n * n; idx += 256) {
      int r = idx / n, c = idx % n;
      if (r == p) continue;
      float fr = Fr[r], fi = Fi[r];
      float apr = Ar[p * n + c], api = Ai[p * n + c];
      Ar[idx] -= fr * apr - fi * api;
      Ai[idx] -= fr * api + fi * apr;
      float mpr = Mr[p * n + c], mpi = Mi[p * n + c];
      Mr[idx] -= fr * mpr - fi * mpi;
      Mi[idx] -= fr * mpi + fi * mpr;
    }
    __syncthreads();
  }
}

// ---------------------------------------------------------------------------
// Main persistent kernel: 64 blocks x 16 rows; loops all T=256 steps.
// Per-row scan state lives in LDS the whole time. All global pointers are
// laundered through an opaque zero offset once per step so loop-invariant
// weight/bias loads are NOT hoisted out of the scan loop (which previously
// caused the compiler to preload every B-fragment and spill them to private
// scratch -> per-wave scratch reloads with zero cross-wave reuse, instead of
// shared L2/WGP$-resident loads from the packed weight buffers).
// ---------------------------------------------------------------------------
__global__ __launch_bounds__(256) void s5_main_kernel(
    const float* __restrict__ z_dyn, const float* __restrict__ z_static,
    const float* __restrict__ dtv, const float* __restrict__ U,
    const float* __restrict__ b0f, const float* __restrict__ b1f,
    const float* __restrict__ sel_b2, const float* __restrict__ alpha_b,
    const float* __restrict__ omega_b, const float* __restrict__ Btr_b,
    const float* __restrict__ Bti_b, const float* __restrict__ Ctr_b,
    const float* __restrict__ Cti_b, const float* __restrict__ Dt_b,
    const bf16* __restrict__ W0p, const bf16* __restrict__ W1p,
    const bf16* __restrict__ W2p, const bf16* __restrict__ Awp,
    const bf16* __restrict__ Owp, const bf16* __restrict__ Brp,
    const bf16* __restrict__ Bip, const bf16* __restrict__ Crp,
    const bf16* __restrict__ Cip, const bf16* __restrict__ Dtp,
    const bf16* __restrict__ Pinvrp, const bf16* __restrict__ Pinvip,
    const bf16* __restrict__ Vtrp, const bf16* __restrict__ Vtip,
    float* __restrict__ Zout, float* __restrict__ Yout) {
  const int tid = threadIdx.x;
  const int lane = tid & 31;
  const int wave = tid >> 5;
  const int row0 = blockIdx.x * 16;
  const int mo = (lane & 16) ? 8 : 0;
  const int u16 = lane & 15;

  // LDS pool, 56KB, phase-aliased.
  __shared__ alignas(64) unsigned char smem[57344];
  bf16*  hz    = (bf16*)(smem);                  // [16][384] bf16  (12288 B)
  bf16*  xa    = (bf16*)(smem + 12288);          // [16][416] bf16  } alias 16KB
  float* lam_r = (float*)(smem + 12288);         // [16][128] f32   }
  float* lam_i = (float*)(smem + 12288 + 8192);  // [16][128] f32   }
  bf16*  h0    = (bf16*)(smem + 28672);          // [16][128] bf16  } alias 4KB
  float* udt   = (float*)(smem + 28672);         // [16][16]  f32   }
  float* yacc  = (float*)(smem + 28672 + 1024);  // [16][32]  f32   }
  float* wf_r  = (float*)(smem + 32768);         // [16][128] f32: w, then z_next
  float* wf_i  = (float*)(smem + 40960);         // [16][128] f32
  bf16*  zb_r  = (bf16*)(smem + 49152);          // [16][128] bf16 (z, then Lam*w)
  bf16*  zb_i  = (bf16*)(smem + 53248);          // [16][128] bf16

  // init state z from z_dyn (real | imag halves)
  for (int i = tid; i < 16 * 128; i += 256) {
    int m = i >> 7, c = i & 127;
    wf_r[i] = z_dyn[(size_t)(row0 + m) * 256 + c];
    wf_i[i] = z_dyn[(size_t)(row0 + m) * 256 + 128 + c];
  }
  __syncthreads();

#pragma unroll 1
  for (int t = 0; t < T_; ++t) {
    // Opaque zero, re-materialized every step: blocks LICM/spill of the
    // t-invariant weight & bias streams.
    size_t zo = 0;
    asm volatile("" : "+s"(zo));
    const bf16* W0q = W0p + zo;   const bf16* W1q = W1p + zo;
    const bf16* W2q = W2p + zo;   const bf16* Awq = Awp + zo;
    const bf16* Owq = Owp + zo;   const bf16* Brq = Brp + zo;
    const bf16* Biq = Bip + zo;   const bf16* Crq = Crp + zo;
    const bf16* Ciq = Cip + zo;   const bf16* Dtq = Dtp + zo;
    const bf16* Pirq = Pinvrp + zo; const bf16* Piiq = Pinvip + zo;
    const bf16* Vtrq = Vtrp + zo; const bf16* Vtiq = Vtip + zo;
    const float* zstatq = z_static + zo; const float* Uq = U + zo;
    const float* dtq = dtv + zo;
    const float* b0q = b0f + zo;  const float* b1q = b1f + zo;
    const float* b2q = sel_b2 + zo;
    const float* abq = alpha_b + zo; const float* obq = omega_b + zo;
    const float* Brbq = Btr_b + zo;  const float* Bibq = Bti_b + zo;
    const float* Crbq = Ctr_b + zo;  const float* Cibq = Cti_b + zo;
    const float* Dbq = Dt_b + zo;

    // ---- Phase A: build selector input tile + bf16 copy of z ----
    for (int i = tid; i < 16 * 416; i += 256) {
      int m = i / 416, c = i % 416;
      float v;
      if (c < 128)      v = wf_r[m * 128 + c];
      else if (c < 256) v = wf_i[m * 128 + (c - 128)];
      else if (c < 384) v = zstatq[(size_t)(row0 + m) * 128 + (c - 256)];
      else if (c < 400) v = Uq[((size_t)t * B_ + row0 + m) * UD_ + (c - 384)];
      else if (c == 400) v = dtq[row0 + m];
      else v = 0.0f;
      xa[i] = (bf16)v;
    }
    for (int i = tid; i < 16 * 128; i += 256) {
      zb_r[i] = (bf16)wf_r[i];
      zb_i[i] = (bf16)wf_i[i];
    }
    __syncthreads();

    // ---- Selector layer 0: [16x416] x [416x128], BN folded, ReLU ----
    {
      v8f acc = vzero8();
      for (int kt = 0; kt < 13; ++kt)
        acc = wmma_bf16(a_frag(xa, 416, kt * 32, lane),
                        b_frag(W0q, kt, wave, 8, lane), acc);
      float bv = b0q[wave * 16 + u16];
#pragma unroll
      for (int r = 0; r < 8; ++r) {
        float v = acc[r] + bv;
        v = v > 0.0f ? v : 0.0f;
        h0[(r + mo) * 128 + wave * 16 + u16] = (bf16)v;
      }
    }
    __syncthreads();

    // ---- Selector layer 1: [16x128] x [128x128], ReLU (in-place h buffer) ----
    {
      v8f acc = vzero8();
      for (int kt = 0; kt < 4; ++kt)
        acc = wmma_bf16(a_frag(h0, 128, kt * 32, lane),
                        b_frag(W1q, kt, wave, 8, lane), acc);
      __syncthreads();  // all reads of h0 done before overwrite
      float bv = b1q[wave * 16 + u16];
#pragma unroll
      for (int r = 0; r < 8; ++r) {
        float v = acc[r] + bv;
        v = v > 0.0f ? v : 0.0f;
        h0[(r + mo) * 128 + wave * 16 + u16] = (bf16)v;
      }
    }
    __syncthreads();

    // ---- Selector layer 2: [16x128] x [128x384] -> hz ----
    for (int c3 = 0; c3 < 3; ++c3) {
      int nt = wave + 8 * c3;
      v8f acc = vzero8();
      for (int kt = 0; kt < 4; ++kt)
        acc = wmma_bf16(a_frag(h0, 128, kt * 32, lane),
                        b_frag(W2q, kt, nt, 24, lane), acc);
      float bv = b2q[nt * 16 + u16];
#pragma unroll
      for (int r = 0; r < 8; ++r)
        hz[(r + mo) * 384 + nt * 16 + u16] = (bf16)(acc[r] + bv);
    }
    __syncthreads();

    // ---- u*dt tile, y accumulator (reuses dead h buffer) + alpha/omega ----
    {
      int m = tid >> 4, u = tid & 15;
      udt[tid] = Uq[((size_t)t * B_ + row0 + m) * UD_ + u] * dtq[row0 + m];
    }
    for (int i = tid; i < 512; i += 256) yacc[i] = 0.0f;
    {
      // |Lam| = exp(-softplus(a)) = 1/(1+e^a)
      v8f acc = vzero8();
      for (int kt = 0; kt < 12; ++kt)
        acc = wmma_bf16(a_frag(hz, 384, kt * 32, lane),
                        b_frag(Awq, kt, wave, 8, lane), acc);
      float bv = abq[wave * 16 + u16];
#pragma unroll
      for (int r = 0; r < 8; ++r) {
        float a = acc[r] + bv;
        lam_r[(r + mo) * 128 + wave * 16 + u16] = 1.0f / (1.0f + __expf(a));
      }
      v8f ac2 = vzero8();
      for (int kt = 0; kt < 12; ++kt)
        ac2 = wmma_bf16(a_frag(hz, 384, kt * 32, lane),
                        b_frag(Owq, kt, wave, 8, lane), ac2);
      float bo = obq[wave * 16 + u16];
#pragma unroll
      for (int r = 0; r < 8; ++r)
        lam_i[(r + mo) * 128 + wave * 16 + u16] = ac2[r] + bo;
    }
    __syncthreads();
    for (int i = tid; i < 2048; i += 256) {  // Lam = mag * e^{i*omega}
      float mag = lam_r[i], om = lam_i[i];
      float s, c;
      __sincosf(om, &s, &c);
      lam_r[i] = mag * c;
      lam_i[i] = mag * s;
    }
    __syncthreads();

    // ---- Rotation 1: w = z @ Vinv^T (complex, 4 real WMMA chains) ----
    {
      v8f arr = vzero8(), aii = vzero8(), ari = vzero8(), air = vzero8();
      for (int kt = 0; kt < 4; ++kt) {
        v16bf azr = a_frag(zb_r, 128, kt * 32, lane);
        v16bf azi = a_frag(zb_i, 128, kt * 32, lane);
        v16bf bpr = b_frag(Pirq, kt, wave, 8, lane);
        v16bf bpi = b_frag(Piiq, kt, wave, 8, lane);
        arr = wmma_bf16(azr, bpr, arr);
        aii = wmma_bf16(azi, bpi, aii);
        ari = wmma_bf16(azr, bpi, ari);
        air = wmma_bf16(azi, bpr, air);
      }
      int n = wave * 16 + u16;
#pragma unroll
      for (int r = 0; r < 8; ++r) {
        wf_r[(r + mo) * 128 + n] = arr[r] - aii[r];
        wf_i[(r + mo) * 128 + n] = ari[r] + air[r];
      }
    }
    __syncthreads();
    // ---- Lam * w -> bf16 (reuse zb buffers; zb dead after rot1) ----
    for (int i = tid; i < 2048; i += 256) {
      float wr = wf_r[i], wi = wf_i[i];
      float lr = lam_r[i], li = lam_i[i];
      zb_r[i] = (bf16)(lr * wr - li * wi);
      zb_i[i] = (bf16)(lr * wi + li * wr);
    }
    __syncthreads();
    // ---- Rotation 2: z_rot = (Lam*w) @ V^T -> wf (becomes z_next accum) ----
    {
      v8f arr = vzero8(), aii = vzero8(), ari = vzero8(), air = vzero8();
      for (int kt = 0; kt < 4; ++kt) {
        v16bf avr = a_frag(zb_r, 128, kt * 32, lane);
        v16bf avi = a_frag(zb_i, 128, kt * 32, lane);
        v16bf bvr = b_frag(Vtrq, kt, wave, 8, lane);
        v16bf bvi = b_frag(Vtiq, kt, wave, 8, lane);
        arr = wmma_bf16(avr, bvr, arr);
        aii = wmma_bf16(avi, bvi, aii);
        ari = wmma_bf16(avr, bvi, ari);
        air = wmma_bf16(avi, bvr, air);
      }
      int n = wave * 16 + u16;
#pragma unroll
      for (int r = 0; r < 8; ++r) {
        wf_r[(r + mo) * 128 + n] = arr[r] - aii[r];
        wf_i[(r + mo) * 128 + n] = ari[r] + air[r];
      }
    }
    __syncthreads();

    // ---- Bt einsum: z_next += (hz @ Btr/Bti^T + b).reshape(d,u) @ (u*dt) ----
    // Each 16x16 output tile of the Bt GEMM is exactly one d (16 u-columns);
    // reduce over u with shfl_xor inside the 16-lane group.
#pragma unroll 1
    for (int j = 0; j < 16; ++j) {
      int d = wave * 16 + j;
      if (j + 1 < 16) {  // cover L2 latency for the next tile's fragments
        __builtin_prefetch(Brq + (((size_t)(d + 1)) * 32 + lane) * 16, 0, 0);
        __builtin_prefetch(Biq + (((size_t)(d + 1)) * 32 + lane) * 16, 0, 0);
      }
      v8f ar = vzero8(), ai = vzero8();
      for (int kt = 0; kt < 12; ++kt) {
        v16bf a = a_frag(hz, 384, kt * 32, lane);
        ar = wmma_bf16(a, b_frag(Brq, kt, d, 128, lane), ar);
        ai = wmma_bf16(a, b_frag(Biq, kt, d, 128, lane), ai);
      }
      float bbr = Brbq[d * 16 + u16];
      float bbi = Bibq[d * 16 + u16];
#pragma unroll
      for (int r = 0; r < 8; ++r) {
        float uv = udt[(r + mo) * 16 + u16];
        float vr = (ar[r] + bbr) * uv;
        float vi = (ai[r] + bbi) * uv;
        for (int off = 1; off < 16; off <<= 1) {
          vr += __shfl_xor(vr, off, 32);
          vi += __shfl_xor(vi, off, 32);
        }
        if (u16 == 0) {
          wf_r[(r + mo) * 128 + d] += vr;
          wf_i[(r + mo) * 128 + d] += vi;
        }
      }
    }
    __syncthreads();

    // ---- Write Z[t] (zr_next = [real | imag]) ----
    for (int i = tid; i < 16 * 256; i += 256) {
      int m = i >> 8, c = i & 255;
      float v = (c < 128) ? wf_r[m * 128 + c] : wf_i[m * 128 + (c - 128)];
      Zout[((size_t)t * B_ + row0 + m) * 256 + c] = v;
    }

    // ---- y = Re(Ct z_next) + Dt (u*dt): 200 Ct tile-jobs + 25 Dt jobs ----
#pragma unroll 1
    for (int job = wave; job < 225; job += 8) {
      if (job < 200) {
        int n = job >> 3, dt8 = job & 7;
        int nt = n * 8 + dt8;
        if (job + 8 < 200) {
          int nt2 = ((job + 8) >> 3) * 8 + ((job + 8) & 7);
          __builtin_prefetch(Crq + (((size_t)nt2) * 32 + lane) * 16, 0, 0);
          __builtin_prefetch(Ciq + (((size_t)nt2) * 32 + lane) * 16, 0, 0);
        }
        v8f cr = vzero8(), ci = vzero8();
        for (int kt = 0; kt < 12; ++kt) {
          v16bf a = a_frag(hz, 384, kt * 32, lane);
          cr = wmma_bf16(a, b_frag(Crq, kt, nt, 200, lane), cr);
          ci = wmma_bf16(a, b_frag(Ciq, kt, nt, 200, lane), ci);
        }
        int d = dt8 * 16 + u16;
        float cbr = Crbq[n * 128 + d];
        float cbi = Cibq[n * 128 + d];
#pragma unroll
        for (int r = 0; r < 8; ++r) {
          int m = r + mo;
          float v = (cr[r] + cbr) * wf_r[m * 128 + d] -
                    (ci[r] + cbi) * wf_i[m * 128 + d];
          for (int off = 1; off < 16; off <<= 1) v += __shfl_xor(v, off, 32);
          if (u16 == 0) atomicAdd(&yacc[m * 32 + n], v);
        }
      } else {
        int n = job - 200;
        v8f ac = vzero8();
        for (int kt = 0; kt < 12; ++kt)
          ac = wmma_bf16(a_frag(hz, 384, kt * 32, lane),
                         b_frag(Dtq, kt, n, 25, lane), ac);
        float db = Dbq[n * 16 + u16];
#pragma unroll
        for (int r = 0; r < 8; ++r) {
          int m = r + mo;
          float v = (ac[r] + db) * udt[m * 16 + u16];
          for (int off = 1; off < 16; off <<= 1) v += __shfl_xor(v, off, 32);
          if (u16 == 0) atomicAdd(&yacc[m * 32 + n], v);
        }
      }
    }
    __syncthreads();
    for (int i = tid; i < 16 * NOBS_; i += 256) {
      int m = i / NOBS_, n = i % NOBS_;
      Yout[((size_t)t * B_ + row0 + m) * NOBS_ + n] = yacc[m * 32 + n];
    }
    __syncthreads();
  }
}

// ---------------------------------------------------------------------------
extern "C" void kernel_launch(void* const* d_in, const int* in_sizes, int n_in,
                              void* d_out, int out_size, void* d_ws, size_t ws_size,
                              hipStream_t stream) {
  (void)in_sizes; (void)n_in; (void)out_size; (void)ws_size;
  const float* z_dyn    = (const float*)d_in[0];
  const float* z_static = (const float*)d_in[1];
  const float* dtv      = (const float*)d_in[2];
  const float* U        = (const float*)d_in[3];
  const float* sel_w0   = (const float*)d_in[4];
  const float* sel_b0   = (const float*)d_in[5];
  const float* bn0_g    = (const float*)d_in[6];
  const float* bn0_b    = (const float*)d_in[7];
  const float* bn0_m    = (const float*)d_in[8];
  const float* bn0_v    = (const float*)d_in[9];
  const float* sel_w1   = (const float*)d_in[10];
  const float* sel_b1   = (const float*)d_in[11];
  const float* bn1_g    = (const float*)d_in[12];
  const float* bn1_b    = (const float*)d_in[13];
  const float* bn1_m    = (const float*)d_in[14];
  const float* bn1_v    = (const float*)d_in[15];
  const float* sel_w2   = (const float*)d_in[16];
  const float* sel_b2   = (const float*)d_in[17];
  const float* alpha_w  = (const float*)d_in[18];
  const float* alpha_b  = (const float*)d_in[19];
  const float* omega_w  = (const float*)d_in[20];
  const float* omega_b  = (const float*)d_in[21];
  const float* V_real   = (const float*)d_in[22];
  const float* V_imag   = (const float*)d_in[23];
  const float* Btr_w    = (const float*)d_in[24];
  const float* Btr_b    = (const float*)d_in[25];
  const float* Bti_w    = (const float*)d_in[26];
  const float* Bti_b    = (const float*)d_in[27];
  const float* Ctr_w    = (const float*)d_in[28];
  const float* Ctr_b    = (const float*)d_in[29];
  const float* Cti_w    = (const float*)d_in[30];
  const float* Cti_b    = (const float*)d_in[31];
  const float* Dt_w     = (const float*)d_in[32];
  const float* Dt_b     = (const float*)d_in[33];

  char* base = (char*)d_ws;
  size_t off = 0;
  auto carve = [&](size_t bytes) -> void* {
    void* p = base + off;
    off = (off + bytes + 255) & ~(size_t)255;
    return p;
  };
  auto bsz = [](int KT, int NT) -> size_t { return (size_t)KT * NT * 512 * 2; };

  bf16* W0p   = (bf16*)carve(bsz(13, 8));
  bf16* W1p   = (bf16*)carve(bsz(4, 8));
  bf16* W2p   = (bf16*)carve(bsz(4, 24));
  bf16* Awp   = (bf16*)carve(bsz(12, 8));
  bf16* Owp   = (bf16*)carve(bsz(12, 8));
  bf16* Brp   = (bf16*)carve(bsz(12, 128));
  bf16* Bip   = (bf16*)carve(bsz(12, 128));
  bf16* Crp   = (bf16*)carve(bsz(12, 200));
  bf16* Cip   = (bf16*)carve(bsz(12, 200));
  bf16* Dtp   = (bf16*)carve(bsz(12, 25));
  bf16* Vtrp  = (bf16*)carve(bsz(4, 8));
  bf16* Vtip  = (bf16*)carve(bsz(4, 8));
  bf16* Pinvrp = (bf16*)carve(bsz(4, 8));
  bf16* Pinvip = (bf16*)carve(bsz(4, 8));
  float* s0  = (float*)carve(512);
  float* b0f = (float*)carve(512);
  float* s1  = (float*)carve(512);
  float* b1f = (float*)carve(512);
  float* Ar  = (float*)carve(128 * 128 * 4);
  float* Ai  = (float*)carve(128 * 128 * 4);
  float* Mr  = (float*)carve(128 * 128 * 4);
  float* Mi  = (float*)carve(128 * 128 * 4);

  auto pgrid = [](int KT, int NT) -> int {
    int t = KT * NT * 512;
    int b = (t + 255) / 256;
    return b > 4096 ? 4096 : b;
  };

  fold_bn_k<<<1, 128, 0, stream>>>(sel_b0, bn0_g, bn0_b, bn0_m, bn0_v, s0, b0f);
  fold_bn_k<<<1, 128, 0, stream>>>(sel_b1, bn1_g, bn1_b, bn1_m, bn1_v, s1, b1f);

  pack_b_k<<<pgrid(13, 8), 256, 0, stream>>>(sel_w0, s0, W0p, 128, 401, 13, 8);
  pack_b_k<<<pgrid(4, 8), 256, 0, stream>>>(sel_w1, s1, W1p, 128, 128, 4, 8);
  pack_b_k<<<pgrid(4, 24), 256, 0, stream>>>(sel_w2, nullptr, W2p, 384, 128, 4, 24);
  pack_b_k<<<pgrid(12, 8), 256, 0, stream>>>(alpha_w, nullptr, Awp, 128, 384, 12, 8);
  pack_b_k<<<pgrid(12, 8), 256, 0, stream>>>(omega_w, nullptr, Owp, 128, 384, 12, 8);
  pack_b_k<<<pgrid(12, 128), 256, 0, stream>>>(Btr_w, nullptr, Brp, 2048, 384, 12, 128);
  pack_b_k<<<pgrid(12, 128), 256, 0, stream>>>(Bti_w, nullptr, Bip, 2048, 384, 12, 128);
  pack_b_k<<<pgrid(12, 200), 256, 0, stream>>>(Ctr_w, nullptr, Crp, 3200, 384, 12, 200);
  pack_b_k<<<pgrid(12, 200), 256, 0, stream>>>(Cti_w, nullptr, Cip, 3200, 384, 12, 200);
  pack_b_k<<<pgrid(12, 25), 256, 0, stream>>>(Dt_w, nullptr, Dtp, 400, 384, 12, 25);
  pack_b_k<<<pgrid(4, 8), 256, 0, stream>>>(V_real, nullptr, Vtrp, 128, 128, 4, 8);
  pack_b_k<<<pgrid(4, 8), 256, 0, stream>>>(V_imag, nullptr, Vtip, 128, 128, 4, 8);

  invert_k<<<1, 256, 0, stream>>>(V_real, V_imag, Ar, Ai, Mr, Mi);
  pack_b_k<<<pgrid(4, 8), 256, 0, stream>>>(Mr, nullptr, Pinvrp, 128, 128, 4, 8);
  pack_b_k<<<pgrid(4, 8), 256, 0, stream>>>(Mi, nullptr, Pinvip, 128, 128, 4, 8);

  float* Zout = (float*)d_out;
  float* Yout = Zout + (size_t)T_ * B_ * 256;
  s5_main_kernel<<<dim3(B_ / 16), dim3(256), 0, stream>>>(
      z_dyn, z_static, dtv, U, b0f, b1f, sel_b2, alpha_b, omega_b, Btr_b, Bti_b,
      Ctr_b, Cti_b, Dt_b, W0p, W1p, W2p, Awp, Owp, Brp, Bip, Crp, Cip, Dtp,
      Pinvrp, Pinvip, Vtrp, Vtip, Zout, Yout);
}
